// FP8LinearFast_68951404970066
// MI455X (gfx1250) — 8-bit matrix operand order repaired and checked by reading, compiled, not run on hardware
//
#include <hip/hip_runtime.h>


typedef __attribute__((ext_vector_type(16))) int   v16i;
typedef __attribute__((ext_vector_type(8)))  float v8f;
typedef int v4i_g __attribute__((vector_size(16)));

#define MDIM 8192
#define KDIM 4096
#define NDIM 4096

#define FP8MAX 448.0f
#define EPSF   1e-12f

#define BT 128
#define KT 128

#if __has_builtin(__builtin_amdgcn_global_load_async_to_lds_b128)
#define HAVE_ASYNC_LDS 1
#else
#define HAVE_ASYNC_LDS 0
#endif

__device__ __forceinline__ void async_copy16(const unsigned char* g, unsigned char* l) {
#if HAVE_ASYNC_LDS
    __builtin_amdgcn_global_load_async_to_lds_b128(
        (__attribute__((address_space(1))) v4i_g*)g,
        (__attribute__((address_space(3))) v4i_g*)l,
        0, 0);
#else
    uint4 v = *(const uint4*)g;
    *(uint4*)l = v;
#endif
}

__device__ __forceinline__ void wait_async_lds() {
#if HAVE_ASYNC_LDS
#if __has_builtin(__builtin_amdgcn_s_wait_asynccnt)
    __builtin_amdgcn_s_wait_asynccnt(0);
#else
    asm volatile("s_wait_asynccnt 0" ::: "memory");
#endif
#endif
}

__global__ void FP8Lin_amax_kernel(const float* __restrict__ in, size_t n4,
                                   unsigned* __restrict__ amax_bits, int slot) {
    size_t i      = (size_t)blockIdx.x * blockDim.x + threadIdx.x;
    size_t stride = (size_t)gridDim.x * blockDim.x;
    const float4* p = (const float4*)in;
    float m = 0.0f;
    for (; i < n4; i += stride) {
        float4 v = p[i];
        m = fmaxf(m, fmaxf(fmaxf(fabsf(v.x), fabsf(v.y)),
                           fmaxf(fabsf(v.z), fabsf(v.w))));
    }
    for (int off = 16; off > 0; off >>= 1)
        m = fmaxf(m, __shfl_xor(m, off, 32));
    __shared__ float sm[8];
    if ((threadIdx.x & 31) == 0) sm[threadIdx.x >> 5] = m;
    __syncthreads();
    if (threadIdx.x < 32) {
        float v = (threadIdx.x < (blockDim.x >> 5)) ? sm[threadIdx.x] : 0.0f;
        for (int off = 4; off > 0; off >>= 1)
            v = fmaxf(v, __shfl_xor(v, off, 32));
        if (threadIdx.x == 0)
            atomicMax(amax_bits + slot, __float_as_uint(v));
    }
}

__global__ void FP8Lin_init_kernel(unsigned* amax_bits) {
    if (threadIdx.x < 2) amax_bits[threadIdx.x] = 0u;
}

__device__ __forceinline__ unsigned cvt2_fp8(float a, float b) {
    unsigned r;
    asm volatile("v_cvt_pk_fp8_f32 %0, %1, %2" : "=v"(r) : "v"(a), "v"(b));
    return r & 0xffffu;
}

__global__ void FP8Lin_quant_kernel(const float* __restrict__ in,
                                    unsigned char* __restrict__ outq, size_t n4,
                                    const unsigned* __restrict__ amax_bits, int slot) {
    const float amax  = fmaxf(__uint_as_float(amax_bits[slot]), EPSF);
    const float scale = FP8MAX / amax;
    size_t i      = (size_t)blockIdx.x * blockDim.x + threadIdx.x;
    size_t stride = (size_t)gridDim.x * blockDim.x;
    const float4* p = (const float4*)in;
    unsigned*     q = (unsigned*)outq;
    for (; i < n4; i += stride) {
        float4 v = p[i];
        float a = fminf(fmaxf(v.x * scale, -FP8MAX), FP8MAX);
        float b = fminf(fmaxf(v.y * scale, -FP8MAX), FP8MAX);
        float c = fminf(fmaxf(v.z * scale, -FP8MAX), FP8MAX);
        float d = fminf(fmaxf(v.w * scale, -FP8MAX), FP8MAX);
        q[i] = cvt2_fp8(a, b) | (cvt2_fp8(c, d) << 16);
    }
}

__global__ __launch_bounds__(256, 1)
void FP8Lin_gemm_kernel(const unsigned char* __restrict__ xq,
                        const unsigned char* __restrict__ wq,
                        const unsigned* __restrict__ amax_bits,
                        const _Float16* __restrict__ bias,
                        _Float16* __restrict__ out) {
    __shared__ alignas(16) unsigned char sA[2][BT * KT];
    __shared__ alignas(16) unsigned char sB[2][BT * KT];

    const int t    = threadIdx.x;
    const int lane = t & 31;
    const int wave = t >> 5;
    const int mBlk = blockIdx.y * BT;
    const int nBlk = blockIdx.x * BT;

    const int wM = (wave >> 1) * 32;
    const int wN = (wave & 1) * 64;

    const float ax = fmaxf(__uint_as_float(amax_bits[0]), EPSF);
    const float aw = fmaxf(__uint_as_float(amax_bits[1]), EPSF);
    const float descale = (ax / FP8MAX) * (aw / FP8MAX);

    v8f acc[2][4];
#pragma unroll
    for (int i = 0; i < 2; ++i)
#pragma unroll
        for (int j = 0; j < 4; ++j) {
            v8f z = {0.f, 0.f, 0.f, 0.f, 0.f, 0.f, 0.f, 0.f};
            acc[i][j] = z;
        }

    auto load_tiles = [&](int buf, int kb) {
        const unsigned char* ga = xq + (size_t)mBlk * KDIM + (size_t)kb * KT;
        const unsigned char* gb = wq + (size_t)nBlk * KDIM + (size_t)kb * KT;
        const int row0 = t >> 3;
        const int col  = (t & 7) * 16;
#pragma unroll
        for (int i = 0; i < 4; ++i) {
            const int r = row0 + i * 32;
            async_copy16(ga + (size_t)r * KDIM + col, &sA[buf][r * KT + col]);
            async_copy16(gb + (size_t)r * KDIM + col, &sB[buf][r * KT + col]);
        }
    };

    load_tiles(0, 0);

    const int KITERS = KDIM / KT;
    const int rhalf  = lane >> 4;
    const int l15    = lane & 15;

    for (int kb = 0; kb < KITERS; ++kb) {
        const int buf = kb & 1;
        wait_async_lds();
        __syncthreads();
        if (kb + 1 < KITERS) load_tiles(buf ^ 1, kb + 1);

        v16i afrag[2];
#pragma unroll
        for (int mt = 0; mt < 2; ++mt) {
            const unsigned char* base = &sA[buf][(wM + mt * 16 + l15) * KT];
#pragma unroll
            for (int c = 0; c < 8; ++c) {
                const int koff = (c >> 1) * 32 + rhalf * 16 + (c & 1) * 8;
                unsigned long long qv = *(const unsigned long long*)(base + koff);
                afrag[mt][2 * c]     = (int)(unsigned)qv;
                afrag[mt][2 * c + 1] = (int)(unsigned)(qv >> 32);
            }
        }

        v16i bfrag[4];
#pragma unroll
        for (int nt = 0; nt < 4; ++nt) {
            const unsigned char* base = &sB[buf][(wN + nt * 16 + l15) * KT];
#pragma unroll
            for (int c = 0; c < 4; ++c) {
                const int koff = c * 32 + rhalf * 16;
                uint4 qv = *(const uint4*)(base + koff);
                bfrag[nt][4 * c]     = (int)qv.x;
                bfrag[nt][4 * c + 1] = (int)qv.y;
                bfrag[nt][4 * c + 2] = (int)qv.z;
                bfrag[nt][4 * c + 3] = (int)qv.w;
            }
        }

#pragma unroll
        for (int mt = 0; mt < 2; ++mt)
#pragma unroll
            for (int nt = 0; nt < 4; ++nt)
                acc[mt][nt] = __builtin_amdgcn_wmma_f32_16x16x128_fp8_fp8(
                    afrag[mt], bfrag[nt], (short)0, acc[mt][nt], false, false);
    }

#pragma unroll
    for (int mt = 0; mt < 2; ++mt) {
#pragma unroll
        for (int nt = 0; nt < 4; ++nt) {
            const int col = nBlk + wN + nt * 16 + l15;
            const float bcol = (float)bias[col];
#pragma unroll
            for (int r = 0; r < 8; ++r) {
                const int row = mBlk + wM + mt * 16 + r + rhalf * 8;
                const float v = acc[mt][nt][r] * descale + bcol;
                out[(size_t)row * NDIM + col] = (_Float16)v;
            }
        }
    }
}

extern "C" void kernel_launch(void* const* d_in, const int* in_sizes, int n_in,
                              void* d_out, int out_size, void* d_ws, size_t ws_size,
                              hipStream_t stream) {
    const float*    x    = (const float*)d_in[0];
    const float*    w    = (const float*)d_in[1];
    const _Float16* bias = (const _Float16*)d_in[2];
    _Float16*       out  = (_Float16*)d_out;

    unsigned char* ws        = (unsigned char*)d_ws;
    unsigned*      amax_bits = (unsigned*)ws;
    unsigned char* xq        = ws + 256;
    unsigned char* wq        = xq + (size_t)MDIM * KDIM;

    FP8Lin_init_kernel<<<1, 32, 0, stream>>>(amax_bits);

    const size_t nx4 = (size_t)MDIM * KDIM / 4;
    const size_t nw4 = (size_t)NDIM * KDIM / 4;
    FP8Lin_amax_kernel<<<1024, 256, 0, stream>>>(x, nx4, amax_bits, 0);
    FP8Lin_amax_kernel<<<1024, 256, 0, stream>>>(w, nw4, amax_bits, 1);
    FP8Lin_quant_kernel<<<2048, 256, 0, stream>>>(x, xq, nx4, amax_bits, 0);
    FP8Lin_quant_kernel<<<2048, 256, 0, stream>>>(w, wq, nw4, amax_bits, 1);

    dim3 grid(NDIM / BT, MDIM / BT);
    FP8Lin_gemm_kernel<<<grid, 256, 0, stream>>>(xq, wq, amax_bits, bias, out);
}
